// QwenSparseMoeBlock_3023656976451
// MI455X (gfx1250) — compile-verified, matrix operand order checked by reading, not run on hardware
//
#include <hip/hip_runtime.h>
#include <stdint.h>

// ---------------- problem constants ----------------
static constexpr int T_  = 2048;   // B*S tokens
static constexpr int H_  = 2048;
static constexpr int E_  = 8;
static constexpr int I_  = 1408;
static constexpr int IS_ = 5632;

// ---------------- WMMA / vector types ----------------
typedef __attribute__((ext_vector_type(16))) __bf16 v16bf;
typedef __attribute__((ext_vector_type(8)))  float  v8f;
typedef __attribute__((ext_vector_type(4)))  float  v4f;   // clang-native: OK for nontemporal builtins

union FragU { v16bf v; uint4 q[2]; };

__device__ __forceinline__ uint16_t f2bf(float f) {
  union { float f; uint32_t u; } cv; cv.f = f;
  uint32_t u = cv.u;
  return (uint16_t)((u + 0x7FFFu + ((u >> 16) & 1u)) >> 16);  // RNE
}

// LDS tile pitch: 32 used + 8 pad (uint16 units) => 80B rows, 16B aligned.
static constexpr int PITCH = 40;

// Load one 16x32 bf16 fragment (A-layout; B uses identical quarter-K layout
// from a transposed [n][k] tile). Two ds_load_b128 per lane.
__device__ __forceinline__ v16bf load_frag(const uint16_t* tile, int r0, int lane) {
  const int r   = r0 + (lane & 15);
  const int sel = lane >> 4;                 // 0: K quarters 0,2   1: quarters 1,3
  const uint16_t* p = tile + r * PITCH + sel * 8;
  FragU f;
  f.q[0] = *(const uint4*)(p);               // K = sel*8 .. +7
  f.q[1] = *(const uint4*)(p + 16);          // K = 16+sel*8 .. +7
  return f.v;
}

// Async DMA copy of 32 bytes global->LDS (gfx1250 GLOBAL_LOAD_ASYNC_TO_LDS_B128).
// The instruction offset is applied to BOTH the LDS and global address (ISA 08 §4.4),
// so one address pair covers both 16B chunks. Tracked by ASYNCcnt.
__device__ __forceinline__ void async_copy_32B(const void* gptr, void* lptr) {
  uint32_t ldsa = (uint32_t)(uintptr_t)lptr;           // low 32 bits = LDS offset
  uint64_t ga   = (uint64_t)(uintptr_t)gptr;
  asm volatile(
      "global_load_async_to_lds_b128 %0, %1, off\n\t"
      "global_load_async_to_lds_b128 %0, %1, off offset:16"
      :: "v"(ldsa), "v"(ga) : "memory");
}

__device__ __forceinline__ void wait_async() {
  asm volatile("s_wait_asynccnt 0" ::: "memory");
}

// ---------------- small kernels ----------------
__global__ void k_zero(float* __restrict__ p, int n) {
  int i = blockIdx.x * blockDim.x + threadIdx.x;
  if (i < n) p[i] = 0.0f;
}

__global__ void k_cvt_bf16(const float* __restrict__ x, uint16_t* __restrict__ o, int n) {
  int i = blockIdx.x * blockDim.x + threadIdx.x;
  if (i < n) o[i] = f2bf(x[i]);
}

// Router: one wave per token. 8 expert logits + shared gate scalar in one x pass.
__global__ __launch_bounds__(256) void k_router(
    const float* __restrict__ x, const float* __restrict__ rw,
    const float* __restrict__ gsw, float* __restrict__ logits_out,
    float* __restrict__ routing, float* __restrict__ sscale) {
  const int lane = threadIdx.x & 31;
  const int wid  = threadIdx.x >> 5;
  const int t    = blockIdx.x * 8 + wid;

  float acc[8] = {0.f, 0.f, 0.f, 0.f, 0.f, 0.f, 0.f, 0.f};
  float accS = 0.f;
  const float* xr = x + (size_t)t * H_;
  for (int h = lane; h < H_; h += 32) {
    float xv = xr[h];
    const v4f* wr = (const v4f*)(rw + (size_t)h * 8);
    v4f w0 = wr[0], w1 = wr[1];
    acc[0] += xv * w0[0]; acc[1] += xv * w0[1]; acc[2] += xv * w0[2]; acc[3] += xv * w0[3];
    acc[4] += xv * w1[0]; acc[5] += xv * w1[1]; acc[6] += xv * w1[2]; acc[7] += xv * w1[3];
    accS += xv * gsw[h];
  }
#pragma unroll
  for (int off = 16; off > 0; off >>= 1) {
#pragma unroll
    for (int e = 0; e < 8; ++e) acc[e] += __shfl_xor(acc[e], off, 32);
    accS += __shfl_xor(accS, off, 32);
  }
  if (lane == 0) {
    float mx = acc[0];
#pragma unroll
    for (int e = 1; e < 8; ++e) mx = fmaxf(mx, acc[e]);
    float ex[8];
#pragma unroll
    for (int e = 0; e < 8; ++e) ex[e] = __expf(acc[e] - mx);
    int i1 = 0;
#pragma unroll
    for (int e = 1; e < 8; ++e) if (ex[e] > ex[i1]) i1 = e;   // first max on ties
    int i2 = (i1 == 0) ? 1 : 0;
#pragma unroll
    for (int e = 0; e < 8; ++e)
      if (e != i1 && ex[e] > ex[i2]) i2 = e;
    float inv = 1.0f / (ex[i1] + ex[i2]);                     // softmax denom cancels
    float* rr = routing + (size_t)t * 8;
#pragma unroll
    for (int e = 0; e < 8; ++e) rr[e] = 0.0f;
    rr[i1] = ex[i1] * inv;
    rr[i2] = ex[i2] * inv;
    float* lo = logits_out + (size_t)t * 8;
#pragma unroll
    for (int e = 0; e < 8; ++e) lo[e] = acc[e];
    sscale[t] = 1.0f / (1.0f + __expf(-accS));
  }
}

// ---------------- gated WMMA GEMM ----------------
// Hout[m, n] = bf16( (A@Bu)[m,n] * silu((A@Bg)[m,n]) ),  A bf16 [M,K], Bg/Bu f32 [K,N]
__global__ __launch_bounds__(256) void k_gated_gemm(
    const uint16_t* __restrict__ A, int lda,
    const float* __restrict__ Bg, int ldbg,
    const float* __restrict__ Bu, int ldbu,
    uint16_t* __restrict__ Hout, int N, int K) {
  __shared__ alignas(16) uint16_t sA[128 * PITCH];
  __shared__ alignas(16) uint16_t sG[128 * PITCH];   // transposed [n][k]
  __shared__ alignas(16) uint16_t sU[128 * PITCH];

  const int tid  = threadIdx.x;
  const int lane = tid & 31, wid = tid >> 5;
  const int m0 = blockIdx.y * 128, n0 = blockIdx.x * 128;
  const int wM = (wid & 3) * 32, wN = (wid >> 2) * 64;

  v8f accG[2][4], accU[2][4];
#pragma unroll
  for (int m = 0; m < 2; ++m)
#pragma unroll
    for (int n = 0; n < 4; ++n) { accG[m][n] = (v8f)(0.0f); accU[m][n] = (v8f)(0.0f); }

  const int rowA = tid >> 1, segA = (tid & 1) * 16;  // A: 2 threads/row, 16 bf16 each
  const int bl = tid & 31, bw = tid >> 5;            // B: lane->4 cols, wave->4 k's

  for (int k0 = 0; k0 < K; k0 += 32) {
    // ---- A tile: async DMA global->LDS (already bf16, no register round-trip) ----
    async_copy_32B(A + (size_t)(m0 + rowA) * lda + k0 + segA,
                   sA + rowA * PITCH + segA);

    // ---- B tiles: f32 NT load, bf16 convert, transpose into LDS ----
    v4f gv[4], uv[4];
#pragma unroll
    for (int kk = 0; kk < 4; ++kk) {
      const int kb = bw * 4 + kk;
      gv[kk] = __builtin_nontemporal_load(
          (const v4f*)(Bg + (size_t)(k0 + kb) * ldbg + n0 + bl * 4));
      uv[kk] = __builtin_nontemporal_load(
          (const v4f*)(Bu + (size_t)(k0 + kb) * ldbu + n0 + bl * 4));
    }
#pragma unroll
    for (int n = 0; n < 4; ++n) {
      const int nn = bl * 4 + n;
      uint32_t glo = f2bf(gv[0][n]) | ((uint32_t)f2bf(gv[1][n]) << 16);
      uint32_t ghi = f2bf(gv[2][n]) | ((uint32_t)f2bf(gv[3][n]) << 16);
      uint32_t ulo = f2bf(uv[0][n]) | ((uint32_t)f2bf(uv[1][n]) << 16);
      uint32_t uhi = f2bf(uv[2][n]) | ((uint32_t)f2bf(uv[3][n]) << 16);
      *(uint2*)(sG + nn * PITCH + bw * 4) = uint2{glo, ghi};
      *(uint2*)(sU + nn * PITCH + bw * 4) = uint2{ulo, uhi};
    }
    wait_async();
    __syncthreads();

    v16bf afrag[2];
#pragma unroll
    for (int m = 0; m < 2; ++m) afrag[m] = load_frag(sA, wM + m * 16, lane);
#pragma unroll
    for (int n = 0; n < 4; ++n) {
      v16bf bg = load_frag(sG, wN + n * 16, lane);
      v16bf bu = load_frag(sU, wN + n * 16, lane);
#pragma unroll
      for (int m = 0; m < 2; ++m) {
        accG[m][n] = __builtin_amdgcn_wmma_f32_16x16x32_bf16(
            false, afrag[m], false, bg, (short)0, accG[m][n], false, false);
        accU[m][n] = __builtin_amdgcn_wmma_f32_16x16x32_bf16(
            false, afrag[m], false, bu, (short)0, accU[m][n], false, false);
      }
    }
    __syncthreads();
  }

  // epilogue: up * silu(gate) -> bf16
  const int frow = lane & 15, fsel = lane >> 4;
#pragma unroll
  for (int m = 0; m < 2; ++m)
#pragma unroll
    for (int n = 0; n < 4; ++n)
#pragma unroll
      for (int i = 0; i < 8; ++i) {
        float g = accG[m][n][i], u = accU[m][n][i];
        float h = u * g / (1.0f + __expf(-g));
        int row = m0 + wM + m * 16 + i + fsel * 8;
        int col = n0 + wN + n * 16 + frow;
        Hout[(size_t)row * N + col] = f2bf(h);
      }
}

// ---------------- scaled-accumulate WMMA GEMM ----------------
// Out[m, n] += scale[m*sstride] * (A@B)[m,n],  A bf16 [M,K], B f32 [K,ldo]
__global__ __launch_bounds__(256) void k_acc_gemm(
    const uint16_t* __restrict__ A, int lda,
    const float* __restrict__ B, int ldb,
    const float* __restrict__ scale, int sstride,
    float* __restrict__ Out, int ldo, int K) {
  __shared__ alignas(16) uint16_t sA[128 * PITCH];
  __shared__ alignas(16) uint16_t sB[128 * PITCH];   // transposed [n][k]

  const int tid  = threadIdx.x;
  const int lane = tid & 31, wid = tid >> 5;
  const int m0 = blockIdx.y * 128, n0 = blockIdx.x * 128;
  const int wM = (wid & 3) * 32, wN = (wid >> 2) * 64;

  v8f acc[2][4];
#pragma unroll
  for (int m = 0; m < 2; ++m)
#pragma unroll
    for (int n = 0; n < 4; ++n) acc[m][n] = (v8f)(0.0f);

  const int rowA = tid >> 1, segA = (tid & 1) * 16;
  const int bl = tid & 31, bw = tid >> 5;

  for (int k0 = 0; k0 < K; k0 += 32) {
    // A tile via async DMA (bf16 passthrough)
    async_copy_32B(A + (size_t)(m0 + rowA) * lda + k0 + segA,
                   sA + rowA * PITCH + segA);

    v4f bv[4];
#pragma unroll
    for (int kk = 0; kk < 4; ++kk) {
      const int kb = bw * 4 + kk;
      bv[kk] = __builtin_nontemporal_load(
          (const v4f*)(B + (size_t)(k0 + kb) * ldb + n0 + bl * 4));
    }
#pragma unroll
    for (int n = 0; n < 4; ++n) {
      const int nn = bl * 4 + n;
      uint32_t lo = f2bf(bv[0][n]) | ((uint32_t)f2bf(bv[1][n]) << 16);
      uint32_t hi = f2bf(bv[2][n]) | ((uint32_t)f2bf(bv[3][n]) << 16);
      *(uint2*)(sB + nn * PITCH + bw * 4) = uint2{lo, hi};
    }
    wait_async();
    __syncthreads();

    v16bf afrag[2];
#pragma unroll
    for (int m = 0; m < 2; ++m) afrag[m] = load_frag(sA, wM + m * 16, lane);
#pragma unroll
    for (int n = 0; n < 4; ++n) {
      v16bf bfrag = load_frag(sB, wN + n * 16, lane);
#pragma unroll
      for (int m = 0; m < 2; ++m)
        acc[m][n] = __builtin_amdgcn_wmma_f32_16x16x32_bf16(
            false, afrag[m], false, bfrag, (short)0, acc[m][n], false, false);
    }
    __syncthreads();
  }

  const int frow = lane & 15, fsel = lane >> 4;
#pragma unroll
  for (int m = 0; m < 2; ++m)
#pragma unroll
    for (int i = 0; i < 8; ++i) {
      int row = m0 + wM + m * 16 + i + fsel * 8;
      float s = scale[(size_t)row * sstride];
#pragma unroll
      for (int n = 0; n < 4; ++n) {
        int col = n0 + wN + n * 16 + frow;
        float* o = Out + (size_t)row * ldo + col;
        *o += s * acc[m][n][i];
      }
    }
}

// ---------------- host launcher ----------------
extern "C" void kernel_launch(void* const* d_in, const int* in_sizes, int n_in,
                              void* d_out, int out_size, void* d_ws, size_t ws_size,
                              hipStream_t stream) {
  (void)in_sizes; (void)n_in; (void)out_size; (void)ws_size;
  const float* x          = (const float*)d_in[0];  // [T,H]
  const float* router_w   = (const float*)d_in[1];  // [H,E]
  const float* gate_up_w  = (const float*)d_in[2];  // [E,H,2I]
  const float* out_w      = (const float*)d_in[3];  // [E,I,H]
  const float* sgw        = (const float*)d_in[4];  // [H,IS]
  const float* siw        = (const float*)d_in[5];  // [H,IS]
  const float* sow        = (const float*)d_in[6];  // [IS,H]
  const float* sgsw       = (const float*)d_in[7];  // [H,1]

  float* out    = (float*)d_out;                    // [T,H]
  float* logits = out + (size_t)T_ * H_;            // [T,E]

  char* ws = (char*)d_ws;
  uint16_t* xb  = (uint16_t*)ws;                                     // T*H bf16
  uint16_t* hid = (uint16_t*)(ws + (size_t)T_ * H_ * 2);             // T*I bf16
  uint16_t* shh = (uint16_t*)(ws + (size_t)T_ * H_ * 2 + (size_t)T_ * I_ * 2);  // T*IS bf16
  float* routing = (float*)((char*)shh + (size_t)T_ * IS_ * 2);      // T*E f32
  float* sscale  = routing + (size_t)T_ * E_;                        // T f32

  const int nTH = T_ * H_;
  k_zero<<<nTH / 256, 256, 0, stream>>>(out, nTH);
  k_cvt_bf16<<<nTH / 256, 256, 0, stream>>>(x, xb, nTH);
  k_router<<<T_ / 8, 256, 0, stream>>>(x, router_w, sgsw, logits, routing, sscale);

  for (int e = 0; e < E_; ++e) {
    const float* bg = gate_up_w + (size_t)e * H_ * 2 * I_;
    k_gated_gemm<<<dim3(I_ / 128, T_ / 128), 256, 0, stream>>>(
        xb, H_, bg, 2 * I_, bg + I_, 2 * I_, hid, I_, H_);
    k_acc_gemm<<<dim3(H_ / 128, T_ / 128), 256, 0, stream>>>(
        hid, I_, out_w + (size_t)e * I_ * H_, H_, routing + e, E_, out, H_, I_);
  }

  // shared expert
  k_gated_gemm<<<dim3(IS_ / 128, T_ / 128), 256, 0, stream>>>(
      xb, H_, sgw, IS_, siw, IS_, shh, IS_, H_);
  k_acc_gemm<<<dim3(H_ / 128, T_ / 128), 256, 0, stream>>>(
      shh, IS_, sow, H_, sscale, 1, out, H_, IS_);
}
